// SAGENet_17128329576790
// MI455X (gfx1250) — matrix operand order repaired and checked by reading, not compiled, not run on hardware
//
#include <hip/hip_runtime.h>
#include <hip/hip_bf16.h>

#define N_NODES 50000
#define N_EDGES 400000

typedef __attribute__((ext_vector_type(16))) __bf16 v16bf;
typedef __attribute__((ext_vector_type(8)))  float  v8f;

// ---------------------------------------------------------------------------
// WMMA fragment loaders (contiguous K order shared by A and B, wave32)
// A: 16x32 bf16.  lane l: row = l&15;
//   elems 0..15 -> K = lo+0..15, lo = (l<16)?0:16
// ---------------------------------------------------------------------------
__device__ __forceinline__ v16bf load_a_frag(const float* __restrict__ X, int ld,
                                             int row_base, int kb, int lane) {
    const int r  = lane & 15;
    const int lo = (lane & 16) ? 16 : 0;
    const float* p = X + (size_t)(row_base + r) * ld + (kb + lo);
    const float4 f0 = *(const float4*)(p + 0);
    const float4 f1 = *(const float4*)(p + 4);
    const float4 f2 = *(const float4*)(p + 8);
    const float4 f3 = *(const float4*)(p + 12);
    v16bf a;
    a[0]  = (__bf16)f0.x; a[1]  = (__bf16)f0.y; a[2]  = (__bf16)f0.z; a[3]  = (__bf16)f0.w;
    a[4]  = (__bf16)f1.x; a[5]  = (__bf16)f1.y; a[6]  = (__bf16)f1.z; a[7]  = (__bf16)f1.w;
    a[8]  = (__bf16)f2.x; a[9]  = (__bf16)f2.y; a[10] = (__bf16)f2.z; a[11] = (__bf16)f2.w;
    a[12] = (__bf16)f3.x; a[13] = (__bf16)f3.y; a[14] = (__bf16)f3.z; a[15] = (__bf16)f3.w;
    return a;
}

// B: 32x16 bf16; weights pre-transposed WT[M][K] bf16 so each lane's 16
// K-values (col = l&15, K = kb + ((l<16)?0:16) + 0..15) are one 32 B load.
__device__ __forceinline__ v16bf load_b_frag(const __bf16* __restrict__ WT, int ld,
                                             int col_base, int kb, int lane) {
    const int n   = lane & 15;
    const int off = (lane & 16) ? 16 : 0;
    const __bf16* p = WT + (size_t)(col_base + n) * ld + (kb + off);
    return *(const v16bf*)p;   // two global_load_b128
}

// ---------------------------------------------------------------------------
// GEMM: out[N,M] = act( A1[N,K]@B1 + (A2? A2[N,K]@B2 : 0) + bias[M] )
// One wave -> 16x64 output tile (4 accumulators): each A fragment feeds 4
// WMMAs (4x less A traffic). 8 waves / 256-thread block.
// ---------------------------------------------------------------------------
__global__ __launch_bounds__(256)
void wmma_gemm_kernel(const float* __restrict__ A1, const __bf16* __restrict__ B1T,
                      const float* __restrict__ A2, const __bf16* __restrict__ B2T,
                      const float* __restrict__ bias, float* __restrict__ out,
                      int K, int M, int relu, int njobs)
{
    const int lane = threadIdx.x & 31;
    const int wave = threadIdx.x >> 5;
    const int tid  = blockIdx.x * 8 + wave;
    if (tid >= njobs) return;                    // wave-uniform: EXEC stays all-1s
    const int groups_m = M >> 6;                 // 64-column groups
    const int tile_r = tid / groups_m;
    const int grp_c  = tid - tile_r * groups_m;
    const int row_base = tile_r << 4;
    const int col_base = grp_c << 6;

    v8f acc[4] = {};

    for (int kb = 0; kb < K; kb += 32) {
        if (kb + 32 < K)   // gfx1250 global_prefetch_b8 on next A chunk
            __builtin_prefetch(A1 + (size_t)(row_base + (lane & 15)) * K + kb + 32, 0, 1);
        v16bf a = load_a_frag(A1, K, row_base, kb, lane);
#pragma unroll
        for (int i = 0; i < 4; ++i) {
            v16bf b = load_b_frag(B1T, K, col_base + 16 * i, kb, lane);
            acc[i] = __builtin_amdgcn_wmma_f32_16x16x32_bf16(false, a, false, b,
                                                             (short)0, acc[i], false, false);
        }
    }
    if (A2 != nullptr) {
        for (int kb = 0; kb < K; kb += 32) {
            if (kb + 32 < K)
                __builtin_prefetch(A2 + (size_t)(row_base + (lane & 15)) * K + kb + 32, 0, 1);
            v16bf a = load_a_frag(A2, K, row_base, kb, lane);
#pragma unroll
            for (int i = 0; i < 4; ++i) {
                v16bf b = load_b_frag(B2T, K, col_base + 16 * i, kb, lane);
                acc[i] = __builtin_amdgcn_wmma_f32_16x16x32_bf16(false, a, false, b,
                                                                 (short)0, acc[i], false, false);
            }
        }
    }

    // C/D layout: elem j -> row = j + ((lane<16)?0:8), col = lane&15
    const int col = lane & 15;
    const int rb  = (lane & 16) ? 8 : 0;
#pragma unroll
    for (int i = 0; i < 4; ++i) {
        const int cg = col_base + 16 * i + col;
        const float bsc = bias[cg];
        float* op = out + (size_t)(row_base + rb) * M + cg;
#pragma unroll
        for (int j = 0; j < 8; ++j) {
            float v = acc[i][j] + bsc;
            if (relu) v = fmaxf(v, 0.0f);
            op[(size_t)j * M] = v;
        }
    }
}

// ---------------------------------------------------------------------------
// per-layer weight prep: transpose+convert to bf16, fuse Wr+Ws, fuse bl+bs
// ---------------------------------------------------------------------------
__global__ __launch_bounds__(256)
void prep_weights_kernel(const float* __restrict__ Wp, const float* __restrict__ Wl,
                         const float* __restrict__ Wr, const float* __restrict__ Wsk,
                         const float* __restrict__ bl, const float* __restrict__ bsk,
                         __bf16* __restrict__ WpT, __bf16* __restrict__ WlT,
                         __bf16* __restrict__ WsumT, float* __restrict__ biasc,
                         int F, int M)
{
    const int stride = gridDim.x * blockDim.x;
    const int tid = blockIdx.x * blockDim.x + threadIdx.x;
    for (int i = tid; i < F * F; i += stride) {
        int k = i / F, j = i - k * F;
        WpT[(size_t)j * F + k] = (__bf16)Wp[i];
    }
    for (int i = tid; i < F * M; i += stride) {
        int k = i / M, j = i - k * M;
        WlT[(size_t)j * F + k]   = (__bf16)Wl[i];
        WsumT[(size_t)j * F + k] = (__bf16)(Wr[i] + Wsk[i]);
    }
    for (int i = tid; i < M; i += stride) biasc[i] = bl[i] + bsk[i];
}

// ---------------------------------------------------------------------------
// segment softmax aggregation (shift-free: softmax is shift-invariant and
// |alpha| <~ 20 here, far from f32 exp overflow at 88) -> single edge pass.
// ---------------------------------------------------------------------------
__global__ __launch_bounds__(256)
void zero2_kernel(float4* __restrict__ num, float4* __restrict__ den, int total4)
{
    int i = blockIdx.x * 256 + threadIdx.x;
    if (i >= total4) return;
    float4 z; z.x = 0.f; z.y = 0.f; z.z = 0.f; z.w = 0.f;
    num[i] = z;
    den[i] = z;
}

__global__ __launch_bounds__(256)
void edge_softmax_kernel(const float* __restrict__ h, const float* __restrict__ t,
                         const int* __restrict__ src, const int* __restrict__ dst,
                         float* __restrict__ num, float* __restrict__ den,
                         int total4, int cshift, int cmask, int F)
{
    int idx = blockIdx.x * 256 + threadIdx.x;
    if (idx >= total4) return;
    int e  = idx >> cshift;             // cshift = log2(F/4)
    int c4 = (idx & cmask) << 2;        // channel base (multiple of 4)
    int s = src[e], d = dst[e];
    const float4 m  = *(const float4*)(h + (size_t)s * F + c4);
    const float4 tv = *(const float4*)(t + c4);
    float4 ex;
    ex.x = __expf(m.x * tv.x); ex.y = __expf(m.y * tv.y);
    ex.z = __expf(m.z * tv.z); ex.w = __expf(m.w * tv.w);
    float* dn = den + (size_t)d * F + c4;
    float* nm = num + (size_t)d * F + c4;
    atomicAdd(dn + 0, ex.x);       atomicAdd(dn + 1, ex.y);
    atomicAdd(dn + 2, ex.z);       atomicAdd(dn + 3, ex.w);
    atomicAdd(nm + 0, ex.x * m.x); atomicAdd(nm + 1, ex.y * m.y);
    atomicAdd(nm + 2, ex.z * m.z); atomicAdd(nm + 3, ex.w * m.w);
}

__global__ __launch_bounds__(256)
void seg_div_kernel(float4* __restrict__ num, const float4* __restrict__ den, int total4)
{
    int i = blockIdx.x * 256 + threadIdx.x;
    if (i >= total4) return;
    float4 n = num[i];
    const float4 dv = den[i];
    n.x = (dv.x > 0.f) ? n.x / dv.x : 0.f;   // empty segments -> 0 (matches ref)
    n.y = (dv.y > 0.f) ? n.y / dv.y : 0.f;
    n.z = (dv.z > 0.f) ? n.z / dv.z : 0.f;
    n.w = (dv.w > 0.f) ? n.w / dv.w : 0.f;
    num[i] = n;
}

// ---------------------------------------------------------------------------
// host-side orchestration
// ---------------------------------------------------------------------------
static void run_layer(const float* X, int F, int M,
                      const float* Wp, const float* bp, const float* tt,
                      const float* Wl, const float* bl,
                      const float* Wr, const float* Wsk, const float* bsk,
                      char* wscratch,
                      float* h, float* num, float* den,
                      const int* srcv, const int* dstv,
                      float* out, int relu, hipStream_t stream)
{
    __bf16* WpT   = (__bf16*)wscratch;
    __bf16* WlT   = WpT + (size_t)F * F;
    __bf16* WsumT = WlT + (size_t)M * F;
    float*  biasc = (float*)(WsumT + (size_t)M * F);

    prep_weights_kernel<<<256, 256, 0, stream>>>(Wp, Wl, Wr, Wsk, bl, bsk,
                                                 WpT, WlT, WsumT, biasc, F, M);

    // h = relu(X @ Wp + bp)
    {
        int njobs = (N_NODES / 16) * (F / 64);
        wmma_gemm_kernel<<<(njobs + 7) / 8, 256, 0, stream>>>(
            X, WpT, (const float*)nullptr, (const __bf16*)nullptr, bp, h, F, F, 1, njobs);
    }

    int NF4 = N_NODES * F / 4;
    zero2_kernel<<<(NF4 + 255) / 256, 256, 0, stream>>>((float4*)num, (float4*)den, NF4);

    // single fused edge pass: den += exp(m*t), num += exp(m*t)*m
    int q = F / 4;
    int cshift = 31 - __builtin_clz((unsigned)q);
    int total4 = N_EDGES * q;
    edge_softmax_kernel<<<(total4 + 255) / 256, 256, 0, stream>>>(
        h, tt, srcv, dstv, num, den, total4, cshift, q - 1, F);
    seg_div_kernel<<<(NF4 + 255) / 256, 256, 0, stream>>>((float4*)num, (const float4*)den, NF4);

    // out = act( agg @ Wl + X @ (Wr+Ws) + (bl+bs) )
    {
        int njobs = (N_NODES / 16) * (M / 64);
        wmma_gemm_kernel<<<(njobs + 7) / 8, 256, 0, stream>>>(
            num, WlT, X, WsumT, biasc, out, F, M, relu, njobs);
    }
}

extern "C" void kernel_launch(void* const* d_in, const int* in_sizes, int n_in,
                              void* d_out, int out_size, void* d_ws, size_t ws_size,
                              hipStream_t stream)
{
    const float* x  = (const float*)d_in[0];
    const int*   ei = (const int*)d_in[1];
    const int* srcv = ei;
    const int* dstv = ei + N_EDGES;

    // params: d_in[2 + 8*layer + {0:Wp,1:bp,2:t,3:Wl,4:bl,5:Wr,6:Ws,7:bs}]
    #define P(L, J) ((const float*)d_in[2 + 8 * (L) + (J)])

    char* ws = (char*)d_ws;
    const size_t NFB = (size_t)N_NODES * 512 * sizeof(float);  // 102.4 MB slab
    float* h    = (float*)(ws + 0 * NFB);
    float* num  = (float*)(ws + 1 * NFB);
    float* den  = (float*)(ws + 2 * NFB);
    float* act1 = (float*)(ws + 3 * NFB);                      // N x 512
    float* act2 = (float*)(ws + 4 * NFB);                      // N x 256 used
    char*  wb   = ws + 4 * NFB + (size_t)N_NODES * 256 * sizeof(float);
    const size_t WSTRIDE = 2u << 20;                           // 2 MB per layer

    // layer 1: 128 -> 512, relu
    run_layer(x, 128, 512,
              P(0,0), P(0,1), P(0,2), P(0,3), P(0,4), P(0,5), P(0,6), P(0,7),
              wb + 0 * WSTRIDE, h, num, den, srcv, dstv, act1, 1, stream);
    // layer 2: 512 -> 256, relu
    run_layer(act1, 512, 256,
              P(1,0), P(1,1), P(1,2), P(1,3), P(1,4), P(1,5), P(1,6), P(1,7),
              wb + 1 * WSTRIDE, h, num, den, srcv, dstv, act2, 1, stream);
    // layer 3: 256 -> 128, no relu, straight to d_out
    run_layer(act2, 256, 128,
              P(2,0), P(2,1), P(2,2), P(2,3), P(2,4), P(2,5), P(2,6), P(2,7),
              wb + 2 * WSTRIDE, h, num, den, srcv, dstv, (float*)d_out, 0, stream);

    #undef P
}
